// ContextEncoder_15066745274857
// MI455X (gfx1250) — compile-verified, matrix operand order checked by reading, not run on hardware
//
#include <hip/hip_runtime.h>
#include <hip/hip_bf16.h>

// ---------------------------------------------------------------------------
// Hierarchical relu-RNN context encoder for gfx1250 (CDNA5, wave32, WMMA).
//   L1: per-sentence RNN over 64 steps (2048 sentences)  -> sent_h [2048,150]
//   L2: context RNN over 2048 sentence hiddens           -> out [150]
// v4: branch-free K-tail — the last K-tile of the input projection reads
// x[268..299] (in-bounds, aligned) and the padded f16 W_ih1 zeroes the
// 20 overlapping entries, so no per-lane guards / exec-mask serialization.
// ---------------------------------------------------------------------------

#define EMBED 300
#define HID   150
#define NSENT 2048
#define TLEN  64
#define KX    320   // EMBED padded to multiple of 32
#define NH    160   // HID padded to multiple of 32 (10 tiles of 16)
#define KTAIL 268   // x base offset of the (shifted) last K tile

typedef __attribute__((ext_vector_type(16))) _Float16 v16h;
typedef __attribute__((ext_vector_type(8)))  _Float16 v8h;
typedef __attribute__((ext_vector_type(8)))  float    v8f;

// ---------------------------------------------------------------------------
// Kernel 0: weight conversion fp32 -> f16 with padding; bias folding.
// Padded k layout for Wih1h: kp in [0,288) -> real k = kp
//                            kp in [288,308) -> zero (overlap with tiles 0..8)
//                            kp in [308,320) -> real k = kp - 20  (288..299)
// so the kernel can read the last x tile from x[268..299] unguarded.
// ---------------------------------------------------------------------------
__global__ void prep_kernel(const float* __restrict__ Wih1,
                            const float* __restrict__ Whh1,
                            const float* __restrict__ bih1,
                            const float* __restrict__ bhh1,
                            const float* __restrict__ Wih2,
                            const float* __restrict__ bih2,
                            const float* __restrict__ bhh2,
                            _Float16* __restrict__ Wih1h,   // [NH][KX]
                            _Float16* __restrict__ Whh1h,   // [NH][NH]
                            _Float16* __restrict__ Wih2h,   // [NH][NH]
                            float* __restrict__ bias1,      // [NH]
                            float* __restrict__ bias2)      // [NH]
{
    const int idx    = blockIdx.x * blockDim.x + threadIdx.x;
    const int stride = gridDim.x * blockDim.x;
    for (int i = idx; i < NH * KX; i += stride) {
        int n = i / KX, kp = i % KX;
        float w = 0.0f;
        if (n < HID) {
            if (kp < 288)       w = Wih1[n * EMBED + kp];
            else if (kp >= 308) w = Wih1[n * EMBED + kp - 20];
            // kp in [288,308): zero (those k already covered by tiles 0..8)
        }
        Wih1h[i] = (_Float16)w;
    }
    for (int i = idx; i < NH * NH; i += stride) {
        int n = i / NH, k = i % NH;
        bool ok = (n < HID && k < HID);
        Whh1h[i] = ok ? (_Float16)Whh1[n * HID + k] : (_Float16)0.0f;
        Wih2h[i] = ok ? (_Float16)Wih2[n * HID + k] : (_Float16)0.0f;
    }
    for (int i = idx; i < NH; i += stride) {
        bias1[i] = (i < HID) ? (bih1[i] + bhh1[i]) : 0.0f;
        bias2[i] = (i < HID) ? (bih2[i] + bhh2[i]) : 0.0f;
    }
}

// Build a v16h fragment from two contiguous groups of 8 halfs:
// p[0..7] -> elems 0..7, p[16..23] -> elems 8..15.
__device__ __forceinline__ v16h load_frag16(const _Float16* p) {
    v8h lo = *(const v8h*)(p);
    v8h hi = *(const v8h*)(p + 16);
    return __builtin_shufflevector(lo, hi, 0, 1, 2, 3, 4, 5, 6, 7,
                                           8, 9, 10, 11, 12, 13, 14, 15);
}

// fp32 -> f16 A fragment from a row pointer (unguarded, aligned).
__device__ __forceinline__ v16h cvt_frag16(const float* p) {
    v16h a;
#pragma unroll
    for (int i = 0; i < 8; ++i) {
        a[i]     = (_Float16)p[i];
        a[i + 8] = (_Float16)p[16 + i];
    }
    return a;
}

// ---------------------------------------------------------------------------
// Kernel 1: fused sentence-level RNN.
// A group = 2 waves sharing 16 sentences and one LDS hidden tile; wave 0
// produces output cols 0..79, wave 1 cols 80..159 (5 N-tiles each).
// 64 blocks x 128 threads = 256 waves.
// ---------------------------------------------------------------------------
__global__ __launch_bounds__(128) void sent_rnn_kernel(
    const float* __restrict__ x,        // [NSENT][TLEN][EMBED]
    const _Float16* __restrict__ Wih,   // [NH][KX]   (rows n, cols kp)
    const _Float16* __restrict__ Whh,   // [NH][NH]
    const float* __restrict__ bias1,    // [NH]
    float* __restrict__ sent_h)         // [NSENT][NH]
{
    constexpr int GROUPS = 2;           // groups (of 2 waves) per block
    constexpr int NT = 5;               // N-tiles per wave
    __shared__ __align__(16) _Float16 h_lds[GROUPS][16 * NH];

    const int wave  = threadIdx.x >> 5;
    const int group = wave >> 1;        // which sentence group
    const int ntb   = (wave & 1) * NT;  // this wave's first N-tile
    const int lane  = threadIdx.x & 31;
    const int row   = lane & 15;  // A row (sentence in tile) / B col (neuron)
    const int hi    = lane >> 4;  // lane half selects K-subgroup
    const int koff  = hi * 8;

    const int s_base = (blockIdx.x * GROUPS + group) * 16;
    const long xrow_base = ((long)(s_base + row)) * TLEN * EMBED;

    // Per-lane weight base pointers; ntl-th fragment = base + ntl*16*stride,
    // folding into constant instruction offsets.
    const _Float16* wih_base = Wih + (ntb * 16 + row) * KX + koff;
    const _Float16* whh_base = Whh + (ntb * 16 + row) * NH + koff;
    const _Float16* hrow = &h_lds[group][row * NH + koff];

    // zero the hidden tiles
    for (int i = threadIdx.x; i < GROUPS * 16 * NH; i += 128)
        ((_Float16*)h_lds)[i] = (_Float16)0.0f;

    float bias_r[NT];
#pragma unroll
    for (int nt = 0; nt < NT; ++nt) bias_r[nt] = bias1[(ntb + nt) * 16 + row];

    __syncthreads();

#pragma unroll 1
    for (int t = 0; t < TLEN; ++t) {
        const float* xrow = x + xrow_base + (long)t * EMBED;
        if (t + 1 < TLEN) __builtin_prefetch(xrow + EMBED, 0, 1);

        v8f acc[NT];
#pragma unroll
        for (int nt = 0; nt < NT; ++nt)
#pragma unroll
            for (int i = 0; i < 8; ++i) acc[nt][i] = bias_r[nt];

        // ---- input projection: x_t @ W_ih^T, K tiles 0..8 ----
#pragma unroll 1
        for (int kt = 0; kt < 9; ++kt) {
            v16h a = cvt_frag16(xrow + kt * 32 + koff);
            const _Float16* w = wih_base + kt * 32;
#pragma unroll
            for (int nt = 0; nt < NT; ++nt) {
                v16h b = load_frag16(w + nt * (16 * KX));
                acc[nt] = __builtin_amdgcn_wmma_f32_16x16x32_f16(
                    false, a, false, b, (short)0, acc[nt], false, false);
            }
        }
        { // K tile 9: x[268..299], overlap zeroed in Wih -> no guards
            v16h a = cvt_frag16(xrow + KTAIL + koff);
            const _Float16* w = wih_base + 288;
#pragma unroll
            for (int nt = 0; nt < NT; ++nt) {
                v16h b = load_frag16(w + nt * (16 * KX));
                acc[nt] = __builtin_amdgcn_wmma_f32_16x16x32_f16(
                    false, a, false, b, (short)0, acc[nt], false, false);
            }
        }

        // ---- recurrence: h_{t-1} @ W_hh^T, K = 160 (5 tiles) ----
#pragma unroll 1
        for (int kt = 0; kt < 5; ++kt) {
            v16h a = load_frag16(hrow + kt * 32);
            const _Float16* w = whh_base + kt * 32;
#pragma unroll
            for (int nt = 0; nt < NT; ++nt) {
                v16h b = load_frag16(w + nt * (16 * NH));
                acc[nt] = __builtin_amdgcn_wmma_f32_16x16x32_f16(
                    false, a, false, b, (short)0, acc[nt], false, false);
            }
        }

        // Both waves of the group finished READING h_{t-1} before overwrite.
        __syncthreads();

        // ---- relu, write hidden tile back (C layout: M = v + 8*hi, N = row)
        const bool last = (t == TLEN - 1);
#pragma unroll
        for (int nt = 0; nt < NT; ++nt) {
#pragma unroll
            for (int v = 0; v < 8; ++v) {
                float r = acc[nt][v];
                r = r > 0.0f ? r : 0.0f;
                int m = v + 8 * hi;
                int n = (ntb + nt) * 16 + row;
                h_lds[group][m * NH + n] = (_Float16)r;
                if (last)
                    sent_h[(long)(s_base + m) * NH + n] = r;
            }
        }
        __syncthreads();  // writes visible before next step's fragment reads
    }
}

// ---------------------------------------------------------------------------
// Kernel 2: context input projection  u2 = sent_h @ W_ih2^T + bias2  (WMMA)
// ---------------------------------------------------------------------------
__global__ __launch_bounds__(128) void ctx_proj_kernel(
    const float* __restrict__ sent_h,   // [NSENT][NH], cols >= HID are zero
    const _Float16* __restrict__ Wih2,  // [NH][NH]
    const float* __restrict__ bias2,    // [NH]
    float* __restrict__ u2)             // [NSENT][NH]
{
    constexpr int WAVES = 4;
    const int wave = threadIdx.x >> 5;
    const int lane = threadIdx.x & 31;
    const int row  = lane & 15;
    const int hi   = lane >> 4;
    const int koff = hi * 8;

    const int m_base = (blockIdx.x * WAVES + wave) * 16;
    const float* arow = sent_h + (long)(m_base + row) * NH;
    const _Float16* w_base = Wih2 + row * NH + koff;

    v8f acc[10];
#pragma unroll
    for (int nt = 0; nt < 10; ++nt) {
        float b = bias2[nt * 16 + row];
#pragma unroll
        for (int i = 0; i < 8; ++i) acc[nt][i] = b;
    }

#pragma unroll 1
    for (int kt = 0; kt < 5; ++kt) {
        v16h a = cvt_frag16(arow + kt * 32 + koff);
        const _Float16* w = w_base + kt * 32;
#pragma unroll
        for (int nt = 0; nt < 10; ++nt) {
            v16h b = load_frag16(w + nt * (16 * NH));
            acc[nt] = __builtin_amdgcn_wmma_f32_16x16x32_f16(
                false, a, false, b, (short)0, acc[nt], false, false);
        }
    }

#pragma unroll
    for (int nt = 0; nt < 10; ++nt)
#pragma unroll
        for (int v = 0; v < 8; ++v) {
            int m = v + 8 * hi;
            u2[(long)(m_base + m) * NH + nt * 16 + row] = acc[nt][v];
        }
}

// ---------------------------------------------------------------------------
// Kernel 3: strictly-sequential context scan (latency-bound; one block).
//   h_t = relu(u2[t] + h_{t-1} @ W_hh2^T)
// Each thread owns one output neuron; its W_hh2 row lives in VGPRs (152),
// h is read as float4 chunks from LDS, 4 independent partial sums for ILP.
// ---------------------------------------------------------------------------
#define KPAD 152   // HID rounded up to multiple of 4

__global__ __launch_bounds__(160) void ctx_scan_kernel(
    const float* __restrict__ u2,     // [NSENT][NH]
    const float* __restrict__ Whh2,   // [HID][HID] original fp32
    float* __restrict__ out)          // [HID]
{
    __shared__ __align__(16) float h[NH];
    const int n = threadIdx.x;
    h[n] = 0.0f;

    // Register-resident weight row (row 0 for the pad threads; masked later).
    const float* wr = Whh2 + (n < HID ? n : 0) * HID;
    float wreg[KPAD];
#pragma unroll
    for (int i = 0; i < KPAD; ++i)
        wreg[i] = (i < HID) ? wr[i] : 0.0f;

    __syncthreads();

#pragma unroll 1
    for (int t = 0; t < NSENT; ++t) {
        float a0 = u2[t * NH + n], a1 = 0.0f, a2 = 0.0f, a3 = 0.0f;
#pragma unroll
        for (int c = 0; c < KPAD / 4; ++c) {
            float4 hv = *(const float4*)&h[c * 4];   // h[150..151] stay 0
            a0 = fmaf(hv.x, wreg[c * 4 + 0], a0);
            a1 = fmaf(hv.y, wreg[c * 4 + 1], a1);
            a2 = fmaf(hv.z, wreg[c * 4 + 2], a2);
            a3 = fmaf(hv.w, wreg[c * 4 + 3], a3);
        }
        float acc = (a0 + a1) + (a2 + a3);
        float hn = (n < HID && acc > 0.0f) ? acc : 0.0f;
        __syncthreads();          // all reads of h_{t-1} done
        h[n] = hn;
        __syncthreads();          // h_t visible
    }
    if (n < HID) out[n] = h[n];
}

// ---------------------------------------------------------------------------
extern "C" void kernel_launch(void* const* d_in, const int* in_sizes, int n_in,
                              void* d_out, int out_size, void* d_ws, size_t ws_size,
                              hipStream_t stream) {
    const float* x     = (const float*)d_in[0];   // [2048,64,300]
    const float* Wih1  = (const float*)d_in[1];   // [150,300]
    const float* Whh1  = (const float*)d_in[2];   // [150,150]
    const float* bih1  = (const float*)d_in[3];   // [150]
    const float* bhh1  = (const float*)d_in[4];   // [150]
    const float* Wih2  = (const float*)d_in[5];   // [150,150]
    const float* Whh2  = (const float*)d_in[6];   // [150,150]
    const float* bih2  = (const float*)d_in[7];   // [150]
    const float* bhh2  = (const float*)d_in[8];   // [150]
    float* out = (float*)d_out;                   // [150]

    char* ws = (char*)d_ws;
    _Float16* Wih1h = (_Float16*)(ws);             // 160*320*2 = 102400 B
    _Float16* Whh1h = (_Float16*)(ws + 102400);    // 160*160*2 =  51200 B
    _Float16* Wih2h = (_Float16*)(ws + 153600);    //              51200 B
    float*    bias1 = (float*)  (ws + 204800);     // 160*4
    float*    bias2 = (float*)  (ws + 205440);     // 160*4
    float*   sent_h = (float*)  (ws + 206080);     // 2048*160*4 = 1310720 B
    float*       u2 = (float*)  (ws + 1516800);    //             1310720 B
    // total ws usage: 2,827,520 bytes

    prep_kernel<<<64, 256, 0, stream>>>(Wih1, Whh1, bih1, bhh1, Wih2,
                                        bih2, bhh2,
                                        Wih1h, Whh1h, Wih2h, bias1, bias2);

    // 2048 sentences / (2 groups * 16 sentences) = 64 blocks, 256 waves
    sent_rnn_kernel<<<64, 128, 0, stream>>>(x, Wih1h, Whh1h, bias1, sent_h);

    // 2048 rows / (4 waves * 16 rows) = 32 blocks
    ctx_proj_kernel<<<32, 128, 0, stream>>>(sent_h, Wih2h, bias2, u2);

    ctx_scan_kernel<<<1, 160, 0, stream>>>(u2, Whh2, out);
}
